// SUCCDR_18305150615650
// MI455X (gfx1250) — matrix operand order repaired and checked by reading, not compiled, not run on hardware
//
#include <hip/hip_runtime.h>
#include <hip/hip_bf16.h>
#include <math.h>

// ---------------------------------------------------------------------------
// CDNA5 / gfx1250 implementation of the SUCCDR meta-weight pipeline.
// GEMMs run on v_wmma_f32_16x16x32_bf16 (bf16 A/B in LDS, f32 accumulate).
// Weights are pre-converted to bf16 B^T[N,Ka] so both LDS tiles stage with
// aligned b128 loads/stores; fragments are fetched as aligned ds_load_b128.
// ---------------------------------------------------------------------------

typedef __attribute__((ext_vector_type(16))) __bf16 v16bf;
typedef __attribute__((ext_vector_type(8)))  float  v8f;

__device__ __forceinline__ unsigned short f32_to_bf16(float x) {
    union { float f; unsigned int u; } a; a.f = x;
    unsigned int u = a.u;
    unsigned int r = u + 0x7fffu + ((u >> 16) & 1u);   // round-to-nearest-even
    return (unsigned short)(r >> 16);
}
__device__ __forceinline__ unsigned int pack_bf16(float lo, float hi) {
    return (unsigned int)f32_to_bf16(lo) | ((unsigned int)f32_to_bf16(hi) << 16);
}

struct SrcPtrs { const void* p[4]; };

// ---------------------------------------------------------------------------
// Tiled WMMA GEMM:  C[M,N] = act(A[M,Ka] @ B[Ka,N] + bias)
//   NSRC  > 1 : A = NSRC concatenated f32 column-blocks of width 128
//   ABF16     : A is a single bf16 [M,Ka] array (MLP chain)
//   OUTBF16   : write bf16 (feeds the next GEMM), else f32
//   Bt is pre-converted, pre-transposed bf16 [N,Ka]; bias is f32[N].
// Block = 256 threads = 8 waves; block tile 32(M) x 64(N); K-step 64;
// two v_wmma_f32_16x16x32_bf16 per wave per stage (one barrier pair).
// Requires M%32==0, N%64==0, Ka%64==0 (true at every call site).
// ---------------------------------------------------------------------------
template<int NSRC, bool ABF16, bool TANH, bool OUTBF16>
__global__ __launch_bounds__(256)
void gemm_wmma(SrcPtrs A, const unsigned short* __restrict__ Bt,
               const float* __restrict__ bias, void* __restrict__ Cout,
               int M, int Ka, int N)
{
    // 144-byte row strides => every b128 access is 16B aligned.
    __shared__ __align__(16) unsigned short As[32][72];   // A tile [m][k], k=0..63
    __shared__ __align__(16) unsigned short Bs[64][72];   // B tile [n][k], k=0..63

    const int t    = threadIdx.x;
    const int lane = t & 31;
    const int wave = t >> 5;
    const int hf   = (lane >> 4) & 1;
    const int l16  = lane & 15;

    const int m0 = blockIdx.x * 32;
    const int n0 = blockIdx.y * 64;
    const int wm = (wave >> 2) * 16;    // 0 or 16
    const int wn = (wave & 3) * 16;     // 0,16,32,48

    const int lda = (NSRC > 1) ? 128 : Ka;

    v8f acc = {};

    for (int k0 = 0; k0 < Ka; k0 += 64) {
        // ---- stage A tile (32 m x 64 k): one 16B LDS store / thread
        {
            const int ml = t >> 3;          // 0..31
            const int kk = (t & 7) * 8;     // 0..56
            if (ABF16) {
                const unsigned short* Ab = (const unsigned short*)A.p[0];
                uint4 v = *(const uint4*)(Ab + (size_t)(m0 + ml) * lda + k0 + kk);
                *(uint4*)&As[ml][kk] = v;
            } else {
                const float* Ablk; int kcol;
                if (NSRC > 1) { Ablk = (const float*)A.p[k0 >> 7]; kcol = (k0 & 127) + kk; }
                else          { Ablk = (const float*)A.p[0];       kcol = k0 + kk;         }
                const float* g = Ablk + (size_t)(m0 + ml) * lda + kcol;
                const float4 v0 = *(const float4*)g;
                const float4 v1 = *(const float4*)(g + 4);
                uint4 o;
                o.x = pack_bf16(v0.x, v0.y);
                o.y = pack_bf16(v0.z, v0.w);
                o.z = pack_bf16(v1.x, v1.y);
                o.w = pack_bf16(v1.z, v1.w);
                *(uint4*)&As[ml][kk] = o;
            }
        }
        // ---- stage B tile (64 n x 64 k) from B^T: two 16B loads+stores / thread
        {
            const int nn = t >> 2;          // 0..63
            const int kk = (t & 3) * 16;    // 0,16,32,48
            const unsigned short* bt = Bt + (size_t)(n0 + nn) * Ka + k0 + kk;
            const uint4 v0 = *(const uint4*)bt;
            const uint4 v1 = *(const uint4*)(bt + 8);
            *(uint4*)&Bs[nn][kk]     = v0;
            *(uint4*)&Bs[nn][kk + 8] = v1;
        }
        // ---- prefetch next stage tiles into cache (global_prefetch_b8)
        if (k0 + 64 < Ka) {
            const int k1 = k0 + 64;
            if (t < 32) {
                if (ABF16) {
                    __builtin_prefetch((const unsigned short*)A.p[0] +
                                       (size_t)(m0 + t) * lda + k1, 0, 1);
                } else {
                    const float* An; int kc;
                    if (NSRC > 1) { An = (const float*)A.p[k1 >> 7]; kc = k1 & 127; }
                    else          { An = (const float*)A.p[0];       kc = k1;       }
                    __builtin_prefetch(An + (size_t)(m0 + t) * lda + kc, 0, 1);
                }
            } else if (t < 96) {
                __builtin_prefetch(Bt + (size_t)(n0 + (t - 32)) * Ka + k1, 0, 1);
            }
        }
        __syncthreads();

        // ---- two WMMAs per stage; fragments via aligned ds_load_b128
        //      (ISA 7.12.2: A lane=m, k-halves {0-7,16-23}/{8-15,24-31};
        //       B lane=n, same k-halves as A; both k-contiguous at dword level)
#pragma unroll
        for (int kh = 0; kh < 2; ++kh) {
            const unsigned short* arow = &As[wm + l16][kh * 32];
            const unsigned short* brow = &Bs[wn + l16][kh * 32];
            union Frag { uint4 q[2]; v16bf v; } fa, fb;
            fa.q[0] = *(const uint4*)(arow + hf * 8);
            fa.q[1] = *(const uint4*)(arow + 16 + hf * 8);
            fb.q[0] = *(const uint4*)(brow + hf * 8);
            fb.q[1] = *(const uint4*)(brow + 16 + hf * 8);
            acc = __builtin_amdgcn_wmma_f32_16x16x32_bf16(
                      false, fa.v, false, fb.v, (short)0, acc, false, false);
        }
        __syncthreads();
    }

    // ---- epilogue: bias (+tanh); C/D layout: m = r + 8*hf, n = l16
    const int ng = n0 + wn + l16;
    const float bv = bias[ng];
    const size_t base = (size_t)(m0 + wm + hf * 8) * N + ng;
    if (OUTBF16) {
        unsigned short* p = (unsigned short*)Cout + base;
#pragma unroll
        for (int r = 0; r < 8; ++r) {
            float v = acc[r] + bv;
            if (TANH) v = tanhf(v);
            *p = f32_to_bf16(v);
            p += N;
        }
    } else {
        float* p = (float*)Cout + base;
#pragma unroll
        for (int r = 0; r < 8; ++r) {
            float v = acc[r] + bv;
            if (TANH) v = tanhf(v);
            *p = v;
            p += N;
        }
    }
}

// ---------------------------------------------------------------------------
// f32 weight [Ka,N] -> bf16 transposed [N,Ka] (once per call; writes coalesced).
// ---------------------------------------------------------------------------
__global__ __launch_bounds__(256)
void cvt_bf16_transpose(const float* __restrict__ w, unsigned short* __restrict__ wt,
                        int Ka, int N)
{
    int idx = blockIdx.x * 256 + threadIdx.x;
    if (idx >= Ka * N) return;
    int n = idx / Ka;
    int k = idx - n * Ka;
    wt[idx] = f32_to_bf16(w[(size_t)k * N + n]);
}

// ---------------------------------------------------------------------------
// Edge scatter-adds (segment_sum): 32 lanes x 4 floats per edge.
// ---------------------------------------------------------------------------
__global__ void scatter_add_u(const float* __restrict__ iemb,
                              const int* __restrict__ edge_user,
                              const int* __restrict__ edge_item,
                              float* __restrict__ outp, int E)
{
    int idx = blockIdx.x * blockDim.x + threadIdx.x;
    int e = idx >> 5;
    if (e >= E) return;
    int dg = (idx & 31) * 4;
    int u = edge_user[e], it = edge_item[e];
    const float4 v = *(const float4*)(iemb + (size_t)it * 128 + dg);
    float* dst = outp + (size_t)u * 128 + dg;
    atomicAdd(dst + 0, v.x); atomicAdd(dst + 1, v.y);
    atomicAdd(dst + 2, v.z); atomicAdd(dst + 3, v.w);
}

__global__ void scatter_add_i(const float* __restrict__ uemb_s,
                              const float* __restrict__ u_emb_sp,
                              const int* __restrict__ edge_user,
                              const int* __restrict__ edge_item,
                              float* __restrict__ outp, int E)
{
    int idx = blockIdx.x * blockDim.x + threadIdx.x;
    int e = idx >> 5;
    if (e >= E) return;
    int dg = (idx & 31) * 4;
    int u = edge_user[e], it = edge_item[e];
    const float4 a = *(const float4*)(uemb_s   + (size_t)u * 128 + dg);
    const float4 b = *(const float4*)(u_emb_sp + (size_t)u * 128 + dg);
    float* dst = outp + (size_t)it * 128 + dg;
    atomicAdd(dst + 0, 0.5f * (a.x + b.x)); atomicAdd(dst + 1, 0.5f * (a.y + b.y));
    atomicAdd(dst + 2, 0.5f * (a.z + b.z)); atomicAdd(dst + 3, 0.5f * (a.w + b.w));
}

// ---------------------------------------------------------------------------
// L2-normalize each 1280-wide f32 row in place (eps-guarded).
// ---------------------------------------------------------------------------
__global__ __launch_bounds__(256)
void l2norm_rows(float* __restrict__ x)
{
    __shared__ float red[256];
    float* row = x + (size_t)blockIdx.x * 1280;
    float ss = 0.f;
#pragma unroll
    for (int j = 0; j < 5; ++j) { float v = row[threadIdx.x + j * 256]; ss += v * v; }
    red[threadIdx.x] = ss; __syncthreads();
    for (int s = 128; s > 0; s >>= 1) {
        if (threadIdx.x < s) red[threadIdx.x] += red[threadIdx.x + s];
        __syncthreads();
    }
    const float scale = 1.0f / fmaxf(sqrtf(red[0]), 1e-12f);
#pragma unroll
    for (int j = 0; j < 5; ++j) row[threadIdx.x + j * 256] *= scale;
}

// ---------------------------------------------------------------------------
// Column sums over [M,1280] (meta-bias means); coalesced row sweeps,
// one atomicAdd per (block, column) at the end.
// ---------------------------------------------------------------------------
__global__ __launch_bounds__(256)
void colsum(const float* __restrict__ x, float* __restrict__ sum, int M)
{
    float acc[5] = {0.f, 0.f, 0.f, 0.f, 0.f};
    for (int r = blockIdx.x; r < M; r += gridDim.x) {
        const float* row = x + (size_t)r * 1280;
#pragma unroll
        for (int j = 0; j < 5; ++j) acc[j] += row[threadIdx.x + j * 256];
    }
#pragma unroll
    for (int j = 0; j < 5; ++j) atomicAdd(&sum[threadIdx.x + j * 256], acc[j]);
}

// ---------------------------------------------------------------------------
// softmax over axis=1 (D=128) of metau[U,128,10]; bias = colsum * invM.
// ---------------------------------------------------------------------------
__global__ __launch_bounds__(128)
void softmax_axis_d(float* __restrict__ x, const float* __restrict__ bsum, float invM)
{
    __shared__ float sm[1280];
    __shared__ float mred[10], sred[10];
    const int d = threadIdx.x;
    float* row = x + (size_t)blockIdx.x * 1280;
    float v[10];
#pragma unroll
    for (int k = 0; k < 10; ++k) {
        v[k] = row[d * 10 + k] + bsum[d * 10 + k] * invM;
        sm[d * 10 + k] = v[k];
    }
    __syncthreads();
    if (d < 10) {
        float m = -3.4e38f;
        for (int j = 0; j < 128; ++j) m = fmaxf(m, sm[j * 10 + d]);
        float s = 0.f;
        for (int j = 0; j < 128; ++j) s += expf(sm[j * 10 + d] - m);
        mred[d] = m; sred[d] = 1.0f / s;
    }
    __syncthreads();
#pragma unroll
    for (int k = 0; k < 10; ++k)
        row[d * 10 + k] = expf(v[k] - mred[k]) * sred[k];
}

// ---------------------------------------------------------------------------
// softmax over axis=1 (K=10) of metai[I,10,128]; fully per-thread.
// ---------------------------------------------------------------------------
__global__ __launch_bounds__(128)
void softmax_axis_k(float* __restrict__ x, const float* __restrict__ bsum, float invM)
{
    const int d = threadIdx.x;
    float* row = x + (size_t)blockIdx.x * 1280;
    float v[10]; float m = -3.4e38f;
#pragma unroll
    for (int k = 0; k < 10; ++k) {
        v[k] = row[k * 128 + d] + bsum[k * 128 + d] * invM;
        m = fmaxf(m, v[k]);
    }
    float s = 0.f;
#pragma unroll
    for (int k = 0; k < 10; ++k) { v[k] = expf(v[k] - m); s += v[k]; }
    const float inv = 1.0f / s;
#pragma unroll
    for (int k = 0; k < 10; ++k) row[k * 128 + d] = v[k] * inv;
}

// ---------------------------------------------------------------------------
extern "C" void kernel_launch(void* const* d_in, const int* in_sizes, int n_in,
                              void* d_out, int out_size, void* d_ws, size_t ws_size,
                              hipStream_t stream)
{
    const float* uemb_s   = (const float*)d_in[0];
    const float* iemb_s   = (const float*)d_in[1];
    const float* uemb_t   = (const float*)d_in[2];
    const float* u_emb_sp = (const float*)d_in[3];
    const int*   edge_user = (const int*)d_in[4];
    const int*   edge_item = (const int*)d_in[5];
    const float* Wu = (const float*)d_in[6];
    const float* bu = (const float*)d_in[7];
    const float* Wi = (const float*)d_in[8];
    const float* bi = (const float*)d_in[9];
    const float* m1w1 = (const float*)d_in[10]; const float* m1b1 = (const float*)d_in[11];
    const float* m1w2 = (const float*)d_in[12]; const float* m1b2 = (const float*)d_in[13];
    const float* m1w3 = (const float*)d_in[14]; const float* m1b3 = (const float*)d_in[15];
    const float* m2w1 = (const float*)d_in[16]; const float* m2b1 = (const float*)d_in[17];
    const float* m2w2 = (const float*)d_in[18]; const float* m2b2 = (const float*)d_in[19];
    const float* m2w3 = (const float*)d_in[20]; const float* m2b3 = (const float*)d_in[21];

    const int U = 80000, I = 40000, Dm = 128;
    const int E = in_sizes[4];
    const int DK = 1280;                  // D*K
    const int CH = 2048;                  // MLP row chunk (bf16 h1 chunk = 5.2 MB, L2-resident)

    typedef unsigned short bf16_t;

    // ---- workspace carve-out (~79 MB, 256B-aligned slabs)
    char* ws = (char*)d_ws;
    float*  nb    = (float*) ws; ws += (size_t)U * Dm * 4;      // f32 neighbors (u then i)
    bf16_t* tmpu  = (bf16_t*)ws; ws += (size_t)U * Dm * 2;      // bf16 activations
    bf16_t* tmpi  = (bf16_t*)ws; ws += (size_t)I * Dm * 2;
    bf16_t* h1    = (bf16_t*)ws; ws += (size_t)CH * DK * 2;
    bf16_t* h2    = (bf16_t*)ws; ws += (size_t)CH * 64 * 2;
    float*  bsumu = (float*) ws; ws += 1280 * 4;
    float*  bsumi = (float*) ws; ws += 1280 * 4;
    // bf16 transposed weights B^T[N,Ka]
    bf16_t* wWu  = (bf16_t*)ws; ws += (size_t)512 * 128 * 2;
    bf16_t* wWi  = (bf16_t*)ws; ws += (size_t)256 * 128 * 2;
    bf16_t* w11  = (bf16_t*)ws; ws += (size_t)128 * DK * 2;
    bf16_t* w12  = (bf16_t*)ws; ws += (size_t)DK * 64 * 2;
    bf16_t* w13  = (bf16_t*)ws; ws += (size_t)64 * DK * 2;
    bf16_t* w21  = (bf16_t*)ws; ws += (size_t)128 * DK * 2;
    bf16_t* w22  = (bf16_t*)ws; ws += (size_t)DK * 64 * 2;
    bf16_t* w23  = (bf16_t*)ws; ws += (size_t)64 * DK * 2;

    float* metau = (float*)d_out;                 // [U,128,10] flat
    float* metai = metau + (size_t)U * DK;        // [I,10,128] flat

    // ---- pre-convert + transpose weights to bf16 (tiny, once per call)
    {
        struct { const float* x; bf16_t* y; int Ka, N; } cv[8] = {
            { Wu,   wWu, 512, 128 }, { Wi,   wWi, 256, 128 },
            { m1w1, w11, 128, DK  }, { m1w2, w12, DK, 64   }, { m1w3, w13, 64, DK },
            { m2w1, w21, 128, DK  }, { m2w2, w22, DK, 64   }, { m2w3, w23, 64, DK },
        };
        for (int j = 0; j < 8; ++j) {
            int n = cv[j].Ka * cv[j].N;
            cvt_bf16_transpose<<<(n + 255) / 256, 256, 0, stream>>>(cv[j].x, cv[j].y,
                                                                    cv[j].Ka, cv[j].N);
        }
    }

    const int sc_blocks = (int)(((long long)E * 32 + 255) / 256);

    // ===== user side graph aggregation + concat GEMM (-> bf16 tmpu) =====
    hipMemsetAsync(nb, 0, (size_t)U * Dm * 4, stream);
    scatter_add_u<<<sc_blocks, 256, 0, stream>>>(iemb_s, edge_user, edge_item, nb, E);
    {
        SrcPtrs a; a.p[0] = uemb_s; a.p[1] = uemb_t; a.p[2] = u_emb_sp; a.p[3] = nb;
        gemm_wmma<4, false, false, true><<<dim3(U / 32, Dm / 64), 256, 0, stream>>>(
            a, wWu, bu, tmpu, U, 512, Dm);
    }

    // ===== item side graph aggregation + concat GEMM (-> bf16 tmpi) =====
    hipMemsetAsync(nb, 0, (size_t)I * Dm * 4, stream);
    scatter_add_i<<<sc_blocks, 256, 0, stream>>>(uemb_s, u_emb_sp, edge_user, edge_item, nb, E);
    {
        SrcPtrs a; a.p[0] = iemb_s; a.p[1] = nb; a.p[2] = nullptr; a.p[3] = nullptr;
        gemm_wmma<2, false, false, true><<<dim3(I / 32, Dm / 64), 256, 0, stream>>>(
            a, wWi, bi, tmpi, I, 256, Dm);
    }

    hipMemsetAsync(bsumu, 0, 1280 * 4, stream);
    hipMemsetAsync(bsumi, 0, 1280 * 4, stream);

    // ===== user MLP (chunked; bf16 intermediates stay in L2) =====
    for (int off = 0; off < U; off += CH) {
        const int mc = (U - off) < CH ? (U - off) : CH;       // 2048 or 128, both %32==0
        SrcPtrs a1; a1.p[0] = tmpu + (size_t)off * Dm; a1.p[1] = a1.p[2] = a1.p[3] = nullptr;
        gemm_wmma<1, true, true,  true ><<<dim3(mc / 32, DK / 64), 256, 0, stream>>>(a1, w11, m1b1, h1, mc, Dm, DK);
        SrcPtrs a2; a2.p[0] = h1;
        a2.p[1] = a2.p[2] = a2.p[3] = nullptr;
        gemm_wmma<1, true, true,  true ><<<dim3(mc / 32, 1),       256, 0, stream>>>(a2, w12, m1b2, h2, mc, DK, 64);
        SrcPtrs a3; a3.p[0] = h2;
        a3.p[1] = a3.p[2] = a3.p[3] = nullptr;
        float* outc = metau + (size_t)off * DK;
        gemm_wmma<1, true, false, false><<<dim3(mc / 32, DK / 64), 256, 0, stream>>>(a3, w13, m1b3, outc, mc, 64, DK);
        l2norm_rows<<<mc, 256, 0, stream>>>(outc);
    }

    // ===== item MLP =====
    for (int off = 0; off < I; off += CH) {
        const int mc = (I - off) < CH ? (I - off) : CH;       // 2048 or 1088, both %32==0
        SrcPtrs a1; a1.p[0] = tmpi + (size_t)off * Dm; a1.p[1] = a1.p[2] = a1.p[3] = nullptr;
        gemm_wmma<1, true, true,  true ><<<dim3(mc / 32, DK / 64), 256, 0, stream>>>(a1, w21, m2b1, h1, mc, Dm, DK);
        SrcPtrs a2; a2.p[0] = h1;
        a2.p[1] = a2.p[2] = a2.p[3] = nullptr;
        gemm_wmma<1, true, true,  true ><<<dim3(mc / 32, 1),       256, 0, stream>>>(a2, w22, m2b2, h2, mc, DK, 64);
        SrcPtrs a3; a3.p[0] = h2;
        a3.p[1] = a3.p[2] = a3.p[3] = nullptr;
        float* outc = metai + (size_t)off * DK;
        gemm_wmma<1, true, false, false><<<dim3(mc / 32, DK / 64), 256, 0, stream>>>(a3, w23, m2b3, outc, mc, 64, DK);
        l2norm_rows<<<mc, 256, 0, stream>>>(outc);
    }

    // ===== meta-bias means, then softmax (bias folded as sum * 1/M) =====
    colsum<<<256, 256, 0, stream>>>(metau, bsumu, U);
    colsum<<<256, 256, 0, stream>>>(metai, bsumi, I);

    softmax_axis_d<<<U, 128, 0, stream>>>(metau, bsumu, 1.0f / (float)U);
    softmax_axis_k<<<I, 128, 0, stream>>>(metai, bsumi, 1.0f / (float)I);
}
